// LWMHSA_3788161155682
// MI455X (gfx1250) — compile-verified, matrix operand order checked by reading, not run on hardware
//
#include <hip/hip_runtime.h>

typedef unsigned short u16;
typedef __attribute__((ext_vector_type(16))) __bf16 v16bf;
typedef __attribute__((ext_vector_type(8)))  float   v8f;
typedef __attribute__((ext_vector_type(4)))  unsigned int u32x4;
typedef __attribute__((ext_vector_type(8)))  int          i32x8;
typedef __attribute__((ext_vector_type(4)))  int          i32x4;

#define WS   7
#define SS   3
#define NH   12
#define T    49
#define TP   64      // padded tokens (4 x 16 M-tiles)
#define C    384
#define HD   32
#define HGRID 56
#define NWW   8      // windows per row/col
#define NWIN_PER_B 64
#define NBIAS ((2 * WS - 1) * (2 * WS - 1) * NH)   // 2028

// ---------- helpers ----------

__device__ __forceinline__ u16 f2bf(float f) {
  return __builtin_bit_cast(u16, (__bf16)f);   // native v_cvt on gfx1250, RNE
}

// Build the 16x32 A-tile / 32x16 B-tile per-lane operand:
// halfs {k..k+7} and {k+16..k+23} relative to p (p must be 16B aligned).
__device__ __forceinline__ v16bf ld_tile(const u16* p) {
  union U { uint4 q[2]; v16bf v; } u;
  u.q[0] = *(const uint4*)(p);
  u.q[1] = *(const uint4*)(p + 16);
  return u.v;
}

__device__ __forceinline__ v8f wmma_bf16(v16bf a, v16bf b, v8f c) {
  return __builtin_amdgcn_wmma_f32_16x16x32_bf16(false, a, false, b, (short)0, c,
                                                 false, false);
}

__device__ __forceinline__ int region1d(int g) {          // Swin shift mask region
  return (g < HGRID - WS) ? 0 : ((g < HGRID - SS) ? 1 : 2);
}

// TDM: copy nelem fp32 from global to LDS as a 1-row 2D tile (D# per ISA 8.3/8.4).
// 6-arg builtin form (amdgpu-toolchain / clang-23).
__device__ __forceinline__ void tdm_copy_f32(const void* gsrc, void* lds_dst, int nelem) {
  unsigned long long ga = (unsigned long long)gsrc;
  unsigned la = (unsigned)(unsigned long long)lds_dst;   // low 32 bits = LDS offset
  unsigned N = (unsigned)nelem;
  u32x4 g0;
  g0[0] = 1u;                                            // count = 1 (valid D#)
  g0[1] = la;                                            // lds_addr
  g0[2] = (unsigned)(ga & 0xFFFFFFFFu);                  // global_addr[31:0]
  g0[3] = (unsigned)((ga >> 32) & 0x1FFFFFFu) | (2u << 30); // global_addr[56:32] | type=2
  i32x8 g1;
  g1[0] = 0x00020000;                                    // data_size = 4B
  g1[1] = (int)((N & 0xFFFFu) << 16);                    // tensor_dim0[15:0]
  g1[2] = (int)((N >> 16) | (1u << 16));                 // tensor_dim0[31:16] | tensor_dim1=1
  g1[3] = (int)((N & 0xFFFFu) << 16);                    // tile_dim0 = N
  g1[4] = 1;                                             // tile_dim1 = 1
  g1[5] = (int)N;                                        // tensor_dim0_stride[31:0]
  g1[6] = (int)((N & 0xFFFFu) << 16);                    // stride0[47:32]=0 | stride1[15:0]
  g1[7] = 0;                                             // stride1[47:16]
  i32x4 z4 = {0, 0, 0, 0};                               // groups 2/3 unused (<=2D)
  i32x8 z8 = {0, 0, 0, 0, 0, 0, 0, 0};
  __builtin_amdgcn_tensor_load_to_lds(g0, g1, z4, z4, z8, 0);
}

// ---------- kernel 0: weight transpose + bf16 convert ----------
__global__ void prep_weights_kernel(const float* __restrict__ qkv_w,   // [384][1152]
                                    const float* __restrict__ proj_w,  // [384][384]
                                    u16* __restrict__ qkvT,            // [1152][384]
                                    u16* __restrict__ projT) {         // [384][384] (T)
  int idx = blockIdx.x * blockDim.x + threadIdx.x;
  const int NQ = 3 * C * C;  // 442368
  if (idx < NQ) {
    int n = idx / C, k = idx % C;
    qkvT[idx] = f2bf(qkv_w[k * (3 * C) + n]);
  } else {
    int i = idx - NQ;
    if (i < C * C) {
      int n = i / C, k = i % C;
      projT[i] = f2bf(proj_w[k * C + n]);
    }
  }
}

// ---------- kernel 1: fully fused shifted-window MHSA, one window / block ----------
__global__ __launch_bounds__(384)
void swin_mhsa_kernel(const float* __restrict__ x,          // [32][56][56][384]
                      const float* __restrict__ qkv_b,      // [1152]
                      const float* __restrict__ proj_b,     // [384]
                      const float* __restrict__ bias_table, // [169][12]
                      const u16*   __restrict__ qkvT,       // [1152][384] bf16
                      const u16*   __restrict__ projT,      // [384][384]  bf16
                      float*       __restrict__ out) {      // [32][56][56][384]
  extern __shared__ __align__(16) char smem[];
  u16* xw  = (u16*)smem;            // [64][384]   (later reused as attn_out)
  u16* qls = xw  + TP * C;          // [64][384]   (later reused as P, part 1)
  u16* kls = qls + TP * C;          // [64][384]   (later reused as P, part 2)
  u16* vls = kls + TP * C;          // [12][32][64] v transposed
  u16* pls   = qls;                 // P overlay: [12][64][64]  (96KB over q+k)
  u16* attls = xw;                  // attn output overlay: [64][384]
  float* btab  = (float*)(vls + TP * C);   // [169][12] relative-position bias
  float* qbls  = btab + NBIAS;             // [1152] qkv bias
  float* pbls  = qbls + 3 * C;             // [384]  proj bias

  const int tid   = threadIdx.x;
  const int lane  = tid & 31;
  const int wave  = tid >> 5;       // 0..11
  const int l16   = lane & 15;
  const int lhalf = lane >> 4;      // 0/1
  const int kofs  = lhalf * 8;      // K-split offset for A/B operands

  const int wi  = blockIdx.x;             // 0..2047
  const int bb  = wi / NWIN_PER_B;
  const int w64 = wi % NWIN_PER_B;
  const int wh7 = (w64 / NWW) * WS;
  const int ww7 = (w64 % NWW) * WS;

  // ---- TDM: stage bias tables into LDS (overlapped with Phase A) ----
  if (wave == 0) {
    tdm_copy_f32(bias_table, btab, NBIAS);
    tdm_copy_f32(qkv_b, qbls, 3 * C);
    tdm_copy_f32(proj_b, pbls, C);
  }

  // ---- Phase A: load rolled window, fp32 -> bf16 into LDS ----
  if (tid < C) {
    for (int t = 0; t < TP; ++t) {
      float v = 0.0f;
      if (t < T) {
        int gh = (wh7 + t / WS + SS) % HGRID;
        int gw = (ww7 + t % WS + SS) % HGRID;
        v = x[(((long)bb * HGRID + gh) * HGRID + gw) * C + tid];
      }
      xw[t * C + tid] = f2bf(v);
    }
  }
  if (wave == 0) __builtin_amdgcn_s_wait_tensorcnt(0);
  __syncthreads();

  // ---- Phase B: QKV GEMM (64x384 @ 384x1152), 288 C-tiles over 12 waves ----
  {
    const int mi = wave & 3;
    const int njb = (wave >> 2) * 24;   // 24 N-tiles per wave
    for (int j = 0; j < 24; ++j) {
      const int nj = njb + j;           // 0..71 over 1152 channels
      __builtin_prefetch(qkvT + (nj * 16) * C, 0, 0);
      v8f acc = {};
      for (int kk = 0; kk < C / 32; ++kk) {
        v16bf a = ld_tile(xw + (mi * 16 + l16) * C + kk * 32 + kofs);
        v16bf b = ld_tile(qkvT + (nj * 16 + l16) * C + kk * 32 + kofs);
        acc = wmma_bf16(a, b, acc);
      }
      const int n = nj * 16 + l16;      // channel in [0,1152)
      const float bias = qbls[n];
#pragma unroll
      for (int r = 0; r < 8; ++r) {
        float v = acc[r] + bias;
        int m = mi * 16 + r + lhalf * 8;       // token 0..63
        if (nj < 24) {                         // Q (pre-scaled)
          qls[m * C + n] = f2bf(v * 0.17677669529663689f);
        } else if (nj < 48) {                  // K
          kls[m * C + (n - C)] = f2bf(v);
        } else {                               // V, stored transposed [head][hd][tok]
          int nv = n - 2 * C;
          vls[(nv >> 5) * (HD * TP) + (nv & 31) * TP + m] = f2bf(v);
        }
      }
    }
  }
  __syncthreads();

  // ---- Phase C: attention scores S = q k^T per head (wave == head) ----
  const int h = wave;
  v8f S[4][4];
  {
    v16bf aq[4], bk[4];
#pragma unroll
    for (int mi = 0; mi < 4; ++mi)
      aq[mi] = ld_tile(qls + (mi * 16 + l16) * C + h * HD + kofs);
#pragma unroll
    for (int nj = 0; nj < 4; ++nj)
      bk[nj] = ld_tile(kls + (nj * 16 + l16) * C + h * HD + kofs);
#pragma unroll
    for (int mi = 0; mi < 4; ++mi)
#pragma unroll
      for (int nj = 0; nj < 4; ++nj) {
        v8f z = {};
        S[mi][nj] = wmma_bf16(aq[mi], bk[nj], z);
      }
  }

  // bias (from LDS) + shift mask + key padding
#pragma unroll
  for (int mi = 0; mi < 4; ++mi)
#pragma unroll
    for (int nj = 0; nj < 4; ++nj)
#pragma unroll
      for (int r = 0; r < 8; ++r) {
        int t = mi * 16 + r + lhalf * 8;
        int s = nj * 16 + l16;
        float v = S[mi][nj][r];
        if (s >= T) {
          v = -1e30f;
        } else if (t < T) {
          int tr = t / WS, tc = t % WS, sr = s / WS, sc = s % WS;
          v += btab[((tr - sr + WS - 1) * (2 * WS - 1) + (tc - sc + WS - 1)) * NH + h];
          int regt = 3 * region1d(wh7 + tr) + region1d(ww7 + tc);
          int regs = 3 * region1d(wh7 + sr) + region1d(ww7 + sc);
          if (regt != regs) v += -100.0f;
        }
        S[mi][nj][r] = v;
      }

  // softmax across the 64 (padded) key columns: 4 nj-tiles x 16 lanes per half
#pragma unroll
  for (int mi = 0; mi < 4; ++mi)
#pragma unroll
    for (int r = 0; r < 8; ++r) {
      float mx = fmaxf(fmaxf(S[mi][0][r], S[mi][1][r]), fmaxf(S[mi][2][r], S[mi][3][r]));
#pragma unroll
      for (int off = 1; off < 16; off <<= 1) mx = fmaxf(mx, __shfl_xor(mx, off, 32));
      float sum = 0.0f;
#pragma unroll
      for (int nj = 0; nj < 4; ++nj) {
        float e = __expf(S[mi][nj][r] - mx);
        S[mi][nj][r] = e;
        sum += e;
      }
#pragma unroll
      for (int off = 1; off < 16; off <<= 1) sum += __shfl_xor(sum, off, 32);
      float inv = 1.0f / sum;
#pragma unroll
      for (int nj = 0; nj < 4; ++nj) S[mi][nj][r] *= inv;
    }

  __syncthreads();   // all reads of q/k done -> safe to overlay P
#pragma unroll
  for (int mi = 0; mi < 4; ++mi)
#pragma unroll
    for (int nj = 0; nj < 4; ++nj)
#pragma unroll
      for (int r = 0; r < 8; ++r) {
        int t = mi * 16 + r + lhalf * 8;
        int s = nj * 16 + l16;
        pls[h * (TP * TP) + t * TP + s] = f2bf(S[mi][nj][r]);
      }
  __syncthreads();

  // ---- Phase D: O = P @ V  (64x64 @ 64x32 per head) ----
  {
    v8f O[4][2];
#pragma unroll
    for (int mi = 0; mi < 4; ++mi)
#pragma unroll
      for (int nd = 0; nd < 2; ++nd) O[mi][nd] = (v8f){};
#pragma unroll
    for (int kk = 0; kk < 2; ++kk) {
      v16bf bv[2];
#pragma unroll
      for (int nd = 0; nd < 2; ++nd)
        bv[nd] = ld_tile(vls + h * (HD * TP) + (nd * 16 + l16) * TP + kk * 32 + kofs);
#pragma unroll
      for (int mi = 0; mi < 4; ++mi) {
        v16bf ap = ld_tile(pls + h * (TP * TP) + (mi * 16 + l16) * TP + kk * 32 + kofs);
#pragma unroll
        for (int nd = 0; nd < 2; ++nd) O[mi][nd] = wmma_bf16(ap, bv[nd], O[mi][nd]);
      }
    }
    // write per-head output into attn_out [64][384] (overlay of xw, dead since B)
#pragma unroll
    for (int mi = 0; mi < 4; ++mi)
#pragma unroll
      for (int nd = 0; nd < 2; ++nd)
#pragma unroll
        for (int r = 0; r < 8; ++r) {
          int t  = mi * 16 + r + lhalf * 8;
          int ch = h * HD + nd * 16 + l16;
          attls[t * C + ch] = f2bf(O[mi][nd][r]);
        }
  }
  __syncthreads();

  // ---- Phase E: projection (64x384 @ 384x384) + inverse roll + store ----
  {
    const int mi = wave & 3;
    const int njb = (wave >> 2) * 8;    // 8 N-tiles per wave (24 total over 384)
    for (int j = 0; j < 8; ++j) {
      const int nj = njb + j;
      __builtin_prefetch(projT + (nj * 16) * C, 0, 0);
      v8f acc = {};
      for (int kk = 0; kk < C / 32; ++kk) {
        v16bf a = ld_tile(attls + (mi * 16 + l16) * C + kk * 32 + kofs);
        v16bf b = ld_tile(projT + (nj * 16 + l16) * C + kk * 32 + kofs);
        acc = wmma_bf16(a, b, acc);
      }
      const int n = nj * 16 + l16;
      const float bias = pbls[n];
#pragma unroll
      for (int r = 0; r < 8; ++r) {
        int t = mi * 16 + r + lhalf * 8;
        if (t < T) {
          int gh = (wh7 + t / WS + SS) % HGRID;
          int gw = (ww7 + t % WS + SS) % HGRID;
          out[(((long)bb * HGRID + gh) * HGRID + gw) * C + n] = acc[r] + bias;
        }
      }
    }
  }
}

// ---------- host launcher ----------
extern "C" void kernel_launch(void* const* d_in, const int* in_sizes, int n_in,
                              void* d_out, int out_size, void* d_ws, size_t ws_size,
                              hipStream_t stream) {
  const float* x          = (const float*)d_in[0];
  const float* qkv_w      = (const float*)d_in[1];
  const float* qkv_b      = (const float*)d_in[2];
  const float* proj_w     = (const float*)d_in[3];
  const float* proj_b     = (const float*)d_in[4];
  const float* bias_table = (const float*)d_in[5];
  float* out = (float*)d_out;

  u16* qkvT = (u16*)d_ws;                 // 1152*384 bf16 = 884736 B
  u16* projT = qkvT + 3 * C * C;          // 384*384  bf16 = 294912 B

  {
    int total = 3 * C * C + C * C;        // 589824
    prep_weights_kernel<<<(total + 255) / 256, 256, 0, stream>>>(qkv_w, proj_w, qkvT, projT);
  }
  {
    const int nwin = 32 * NWIN_PER_B;     // 2048 windows
    const size_t shmem = (size_t)4 * TP * C * sizeof(u16)          // 196608
                       + (size_t)(NBIAS + 3 * C + C) * sizeof(float); // 14256
    swin_mhsa_kernel<<<nwin, 384, shmem, stream>>>(x, qkv_b, proj_b, bias_table,
                                                   qkvT, projT, out);
  }
}
